// QuatBitLinear_11158325035345
// MI455X (gfx1250) — matrix operand order repaired and checked by reading, not compiled, not run on hardware
//
#include <hip/hip_runtime.h>
#include <hip/hip_bf16.h>
#include <cstdint>

// ---------------------------------------------------------------------------
// QuatBitLinear for MI455X (gfx1250, wave32, WMMA + async global->LDS).
//   Out[M=16384, N=2048] = X[M, K=2048] @ Wq[N, K]^T   (Wq = quaternion-structured
//   ternary matrix built on device), then (out + bias) * scale * 0.5.
// Workspace layout (needs ~75.6 MB):
//   [0,   256)              : 8 floats of stats (sum, sum|w|) x 4 weights
//   [256, 256+8MB)          : Wq  f16 [2048][2048]  (n-major = pre-transposed)
//   [256+8MB, +64MB)        : Xh  f16 [16384][2048]
// ---------------------------------------------------------------------------

typedef __attribute__((ext_vector_type(16))) _Float16 v16h;
typedef __attribute__((ext_vector_type(8)))  float    v8f;
typedef __attribute__((address_space(3)))    _Float16 lds_f16;

#define C_DIM   512
#define F_DIM   2048
#define WELEMS  (C_DIM * C_DIM)      // 262144 per weight
#define BM      128
#define BN      128
#define BK      64                   // two WMMA K-steps per LDS stage
#define LDH     72                   // padded f16 stride (144B, 16B-aligned)

// ------------------------- stats: sum(w), sum(|w|) -------------------------
__global__ void k_stats(const float* __restrict__ r, const float* __restrict__ i,
                        const float* __restrict__ j, const float* __restrict__ k,
                        float* __restrict__ stats) {
    const float* wsrc[4] = {r, i, j, k};
    const float* w = wsrc[blockIdx.y];
    __shared__ float ss[256];
    __shared__ float sa[256];
    float s = 0.f, a = 0.f;
    for (int idx = blockIdx.x * 256 + threadIdx.x; idx < WELEMS; idx += 256 * 64) {
        float v = w[idx];
        s += v;
        a += fabsf(v);
    }
    ss[threadIdx.x] = s;
    sa[threadIdx.x] = a;
    __syncthreads();
    for (int off = 128; off > 0; off >>= 1) {
        if (threadIdx.x < off) {
            ss[threadIdx.x] += ss[threadIdx.x + off];
            sa[threadIdx.x] += sa[threadIdx.x + off];
        }
        __syncthreads();
    }
    if (threadIdx.x == 0) {
        atomicAdd(&stats[blockIdx.y * 2 + 0], ss[0]);
        atomicAdd(&stats[blockIdx.y * 2 + 1], sa[0]);
    }
}

// ------------- ternarize + build quaternion block matrix (n-major) ----------
// Wq[n][kk]: n = q*512 + oc (output), kk = p*512 + ic (input).
//   q=r: [ +r -i -j -k ]   q=i: [ +i +r +k -j ]
//   q=j: [ +j -k +r +i ]   q=k: [ +k +j -i +r ]
__global__ void k_quant(const float* __restrict__ r, const float* __restrict__ i,
                        const float* __restrict__ j, const float* __restrict__ k,
                        const float* __restrict__ stats, _Float16* __restrict__ Wq) {
    int id = blockIdx.x * 256 + threadIdx.x;
    if (id >= F_DIM * F_DIM) return;
    int n  = id >> 11;
    int kk = id & 2047;
    int q  = n  >> 9, oc = n  & 511;
    int p  = kk >> 9, ic = kk & 511;

    const int   selw[4][4] = {{0,1,2,3},{1,0,3,2},{2,3,0,1},{3,2,1,0}};
    const float sgn [4][4] = {{1.f,-1.f,-1.f,-1.f},
                              {1.f, 1.f, 1.f,-1.f},
                              {1.f,-1.f, 1.f, 1.f},
                              {1.f, 1.f,-1.f, 1.f}};
    int widx = selw[q][p];
    const float* wsel = (widx == 0) ? r : (widx == 1) ? i : (widx == 2) ? j : k;

    const float inv_n = 1.0f / (float)WELEMS;
    float mean = stats[widx * 2 + 0] * inv_n;
    float sc   = fmaxf(stats[widx * 2 + 1] * inv_n, 1e-8f);
    float v    = wsel[oc * C_DIM + ic];
    float t    = rintf(fminf(1.f, fmaxf(-1.f, (v - mean) / sc)));  // {-1,0,1}
    Wq[id] = (_Float16)(sgn[q][p] * t);
}

// ------------------------- X fp32 -> f16 (vector x4) -----------------------
__global__ void k_cvt(const float* __restrict__ x, _Float16* __restrict__ xh, int n4) {
    int id = blockIdx.x * 256 + threadIdx.x;
    if (id >= n4) return;
    float4 v = ((const float4*)x)[id];
    union { _Float16 h[4]; uint2 u; } pk;
    pk.h[0] = (_Float16)v.x; pk.h[1] = (_Float16)v.y;
    pk.h[2] = (_Float16)v.z; pk.h[3] = (_Float16)v.w;
    ((uint2*)xh)[id] = pk.u;
}

// --------------------------- main WMMA GEMM --------------------------------
// 256 threads = 8 waves; waves 4(M) x 2(N); each wave: 32x64 output = 2x4
// tiles of 16x16 -> 16 v_wmma_f32_16x16x32_f16 per BK=64 stage.
// Tiles staged with GLOBAL_LOAD_ASYNC_TO_LDS_B128 (ASYNCcnt), double-buffered:
//   s_wait_asynccnt 0 -> barrier -> issue(next buf) -> compute(cur buf).
__global__ void __launch_bounds__(256)
k_gemm(const _Float16* __restrict__ Xh, const _Float16* __restrict__ Wq,
       const float* __restrict__ bias, const float* __restrict__ stats,
       float* __restrict__ Out) {
    __shared__ _Float16 As[2 * BM * LDH];   // [buf][m][k], padded
    __shared__ _Float16 Bs[2 * BN * LDH];   // [buf][n][k], padded (B pre-transposed)

    const int m0   = blockIdx.y * BM;
    const int n0   = blockIdx.x * BN;
    const int t    = threadIdx.x;
    const int wave = t >> 5;
    const int lane = t & 31;
    const int wm   = (wave >> 1) * 32;
    const int wn   = (wave & 1) * 64;
    const int half = lane >> 4;
    const int lrow = lane & 15;

    const float inv_n = 1.0f / (float)WELEMS;
    float factor = 0.125f * (fmaxf(stats[1] * inv_n, 1e-8f) + fmaxf(stats[3] * inv_n, 1e-8f) +
                             fmaxf(stats[5] * inv_n, 1e-8f) + fmaxf(stats[7] * inv_n, 1e-8f));

    // issue one BK=64 stage: 1024 x 16B chunks per matrix, 4 per thread each.
    auto stage = [&](int buf, int kk) {
#pragma unroll
        for (int s = 0; s < 4; ++s) {
            int c   = t + 256 * s;
            int row = c >> 3;            // 0..127
            int off = (c & 7) * 8;       // f16 offset within row, 16B granules
            unsigned la = (unsigned)(uintptr_t)(lds_f16*)&As[buf * BM * LDH + row * LDH + off];
            unsigned lb = (unsigned)(uintptr_t)(lds_f16*)&Bs[buf * BN * LDH + row * LDH + off];
            const _Float16* ga = &Xh[(size_t)(m0 + row) * F_DIM + kk + off];
            const _Float16* gb = &Wq[(size_t)(n0 + row) * F_DIM + kk + off];
            asm volatile("global_load_async_to_lds_b128 %0, %1, off"
                         :: "v"(la), "v"(ga) : "memory");
            asm volatile("global_load_async_to_lds_b128 %0, %1, off"
                         :: "v"(lb), "v"(gb) : "memory");
        }
    };

    v8f acc[2][4];
#pragma unroll
    for (int a = 0; a < 2; ++a)
#pragma unroll
        for (int b = 0; b < 4; ++b) acc[a][b] = (v8f){0.f,0.f,0.f,0.f,0.f,0.f,0.f,0.f};

    stage(0, 0);   // prologue fill of buffer 0

    for (int k0 = 0; k0 < F_DIM; k0 += BK) {
        const int cur = (k0 / BK) & 1;
        asm volatile("s_wait_asynccnt 0x0" ::: "memory");  // my stage(cur) landed
        __syncthreads();                                    // everyone's landed; prev reads done
        if (k0 + BK < F_DIM) stage(cur ^ 1, k0 + BK);       // overlap fill with compute

        const _Float16* Ab = &As[cur * BM * LDH];
        const _Float16* Bb = &Bs[cur * BN * LDH];
#pragma unroll
        for (int ks = 0; ks < 2; ++ks) {
            const int kb = ks * 32;
            // A frag 16x32: lanes 0-15 M=lrow K{0..7,16..23}; lanes 16-31 K+8
            v16h af[2];
#pragma unroll
            for (int tm = 0; tm < 2; ++tm) {
                int m = wm + tm * 16 + lrow;
                uint4* d = (uint4*)&af[tm];
                d[0] = *(const uint4*)&Ab[m * LDH + kb + half * 8];
                d[1] = *(const uint4*)&Ab[m * LDH + kb + 16 + half * 8];
            }
            // B frag 32x16: lane column N=lrow; lanes 0-15 K{0..7,16..23}; lanes 16-31 K+8
            v16h bf[4];
#pragma unroll
            for (int tn = 0; tn < 4; ++tn) {
                int n = wn + tn * 16 + lrow;
                uint4* d = (uint4*)&bf[tn];
                d[0] = *(const uint4*)&Bb[n * LDH + kb + half * 8];
                d[1] = *(const uint4*)&Bb[n * LDH + kb + 16 + half * 8];
            }
#pragma unroll
            for (int tm = 0; tm < 2; ++tm)
#pragma unroll
                for (int tn = 0; tn < 4; ++tn)
                    acc[tm][tn] = __builtin_amdgcn_wmma_f32_16x16x32_f16(
                        false, af[tm], false, bf[tn], (short)0, acc[tm][tn], false, false);
        }
        __syncthreads();   // done reading cur before it is refilled next round
    }

    // epilogue: C/D layout — lane col N=lrow; VGPR r -> M = r + half*8
#pragma unroll
    for (int tm = 0; tm < 2; ++tm) {
#pragma unroll
        for (int tn = 0; tn < 4; ++tn) {
            int n = n0 + wn + tn * 16 + lrow;
            float bv = bias[n];
            int mb = m0 + wm + tm * 16 + half * 8;
#pragma unroll
            for (int r = 0; r < 8; ++r) {
                Out[(size_t)(mb + r) * F_DIM + n] = (acc[tm][tn][r] + bv) * factor;
            }
        }
    }
}

// ---------------------------------------------------------------------------
extern "C" void kernel_launch(void* const* d_in, const int* in_sizes, int n_in,
                              void* d_out, int out_size, void* d_ws, size_t ws_size,
                              hipStream_t stream) {
    const float* x  = (const float*)d_in[0];
    const float* rw = (const float*)d_in[1];
    const float* iw = (const float*)d_in[2];
    const float* jw = (const float*)d_in[3];
    const float* kw = (const float*)d_in[4];
    const float* bias = (const float*)d_in[5];
    float* out = (float*)d_out;

    const int M = in_sizes[0] / F_DIM;   // 16384

    float*     stats = (float*)d_ws;                                   // 8 floats
    _Float16*  Wq    = (_Float16*)((char*)d_ws + 256);                 // 8 MB
    _Float16*  Xh    = (_Float16*)((char*)d_ws + 256 + (size_t)F_DIM * F_DIM * 2);

    hipMemsetAsync(stats, 0, 64, stream);

    k_stats<<<dim3(64, 4), 256, 0, stream>>>(rw, iw, jw, kw, stats);

    k_quant<<<(F_DIM * F_DIM + 255) / 256, 256, 0, stream>>>(rw, iw, jw, kw, stats, Wq);

    int n4 = (M * F_DIM) / 4;
    k_cvt<<<(n4 + 255) / 256, 256, 0, stream>>>(x, Xh, n4);

    k_gemm<<<dim3(F_DIM / BN, M / BM), 256, 0, stream>>>(Xh, Wq, bias, stats, out);
}
